// CausalMultiHeadAttention_39831526703802
// MI455X (gfx1250) — compile-verified, matrix operand order checked by reading, not run on hardware
//
#include <hip/hip_runtime.h>

typedef __attribute__((ext_vector_type(16))) _Float16 v16h;
typedef __attribute__((ext_vector_type(8)))  _Float16 v8h;
typedef __attribute__((ext_vector_type(8)))  float    v8f;

#define BATCH 2
#define SEQ   2048
#define DMODEL 1024
#define NHEAD 16
#define DK    64

__device__ __forceinline__ v8f wmma_f16(v16h a, v16h b, v8f c) {
    // D = A(16x32 f16) * B(32x16 f16) + C(16x16 f32)
    return __builtin_amdgcn_wmma_f32_16x16x32_f16(false, a, false, b, (short)0, c,
                                                  false, false);
}

// Load 16 f16 halves: p[0..7] and p[16..23] (two b128 loads) — matches the
// 16-bit A/B-matrix VGPR striping (lanes 0-15: K lo-half, lanes 16-31: K hi-half).
__device__ __forceinline__ v16h load_a16_f16(const _Float16* __restrict__ p) {
    v8h lo = *(const v8h*)(p);
    v8h hi = *(const v8h*)(p + 16);
    return __builtin_shufflevector(lo, hi, 0,1,2,3,4,5,6,7,8,9,10,11,12,13,14,15);
}

// ---------------------------------------------------------------------------
// f32 -> f16 elementwise conversion, 8 elements/thread (b128 in, b128 out)
// ---------------------------------------------------------------------------
__global__ __launch_bounds__(256) void cvt_f32_f16(const float* __restrict__ src,
                                                   _Float16* __restrict__ dst,
                                                   int n8) {
    const int i = blockIdx.x * 256 + threadIdx.x;
    if (i >= n8) return;
    float4 a = ((const float4*)src)[2 * i];
    float4 b = ((const float4*)src)[2 * i + 1];
    v8h r;
    r[0]=(_Float16)a.x; r[1]=(_Float16)a.y; r[2]=(_Float16)a.z; r[3]=(_Float16)a.w;
    r[4]=(_Float16)b.x; r[5]=(_Float16)b.y; r[6]=(_Float16)b.z; r[7]=(_Float16)b.w;
    ((v8h*)dst)[i] = r;
}

// ---------------------------------------------------------------------------
// f16 GEMM: C[m,n] = sum_k A[m,k] * W[n,k]   (A:[M,1024], W:[N,1024] row-major)
// One wave computes a 32x64 register tile: 2 A-tiles x 4 B-tiles -> 8 WMMAs
// per 6 operand-tile loads per k-step of 32.
// mode 0 -> f16 out [B,H,S,DK]   (Q, K)
// mode 1 -> f16 out [B,H,DK,S]   (V transposed for the P*V B-matrix reads)
// mode 2 -> f32 out row-major [M,1024]  (final projection)
// ---------------------------------------------------------------------------
__global__ __launch_bounds__(32) void gemm16(const _Float16* __restrict__ A,
                                             const _Float16* __restrict__ W,
                                             _Float16* __restrict__ outh,
                                             float* __restrict__ outf,
                                             int mode) {
    const int lane  = threadIdx.x;
    const int ln    = lane & 15;
    const int kb    = (lane < 16) ? 0 : 8;
    const int rbase = (lane < 16) ? 0 : 8;
    const int m0 = blockIdx.y * 32;
    const int n0 = blockIdx.x * 64;

    const _Float16* arow0 = A + (size_t)(m0 + ln) * DMODEL;
    const _Float16* arow1 = A + (size_t)(m0 + 16 + ln) * DMODEL;
    const _Float16* brow0 = W + (size_t)(n0 + ln) * DMODEL;

    v8f c[2][4];
#pragma unroll
    for (int i = 0; i < 2; ++i)
#pragma unroll
        for (int t = 0; t < 4; ++t) c[i][t] = v8f{};

#pragma unroll 2
    for (int k0 = 0; k0 < DMODEL; k0 += 32) {
        const v16h a0 = load_a16_f16(arow0 + k0 + kb);
        const v16h a1 = load_a16_f16(arow1 + k0 + kb);
#pragma unroll
        for (int t = 0; t < 4; ++t) {
            const v16h bt = load_a16_f16(brow0 + (size_t)t * 16 * DMODEL + k0 + kb);
            c[0][t] = wmma_f16(a0, bt, c[0][t]);
            c[1][t] = wmma_f16(a1, bt, c[1][t]);
        }
    }

#pragma unroll
    for (int i = 0; i < 2; ++i) {
#pragma unroll
        for (int t = 0; t < 4; ++t) {
            const int n = n0 + t * 16 + ln;
            const int h = n >> 6, d = n & 63;
#pragma unroll
            for (int r = 0; r < 8; ++r) {
                const int m    = m0 + i * 16 + rbase + r;   // token index
                const int bidx = m >> 11;                   // m / SEQ
                const int s    = m & 2047;                  // m % SEQ
                if (mode == 0)
                    outh[(((size_t)(bidx * NHEAD + h) * SEQ + s) << 6) + d] =
                        (_Float16)c[i][t][r];
                else if (mode == 1)
                    outh[(((size_t)(bidx * NHEAD + h) * DK + d) << 11) + s] =
                        (_Float16)c[i][t][r];
                else
                    outf[((size_t)m << 10) + n] = c[i][t][r];
            }
        }
    }
}

// ---------------------------------------------------------------------------
// Flash-style causal attention. One wave per (b, h, 16-query tile).
// Q,K: [B,H,S,DK] f16.  VT: [B,H,DK,S] f16.  AO: [B,S,H*DK] f16.
// ---------------------------------------------------------------------------
__global__ __launch_bounds__(32) void attn_kernel(const _Float16* __restrict__ Q,
                                                  const _Float16* __restrict__ K,
                                                  const _Float16* __restrict__ VT,
                                                  _Float16* __restrict__ AO) {
    __shared__ __align__(16) _Float16 Pl[16 * 32];   // P tile staging (C->A transpose)

    const int wid  = blockIdx.x;
    const int qt   = wid & 127;          // S/16 = 128 query tiles
    const int h    = (wid >> 7) & 15;
    const int b    = wid >> 11;
    const int lane = threadIdx.x;
    const int ln    = lane & 15;
    const int kb    = (lane < 16) ? 0 : 8;
    const int rbase = (lane < 16) ? 0 : 8;
    const int q0 = qt * 16;

    const size_t head = (size_t)(b * NHEAD + h) * SEQ;
    const _Float16* Qrow  = Q  + (head + q0 + ln) * DK;
    const _Float16* Khead = K  + head * DK;
    const _Float16* Vhead = VT + (size_t)(b * NHEAD + h) * DK * SEQ;

    // Q as A-matrix for d_k halves [0,32) and [32,64)
    const v16h qa0 = load_a16_f16(Qrow + kb);
    const v16h qa1 = load_a16_f16(Qrow + 32 + kb);

    v8f o[4] = {v8f{}, v8f{}, v8f{}, v8f{}};
    float mi[8], li[8];
#pragma unroll
    for (int r = 0; r < 8; ++r) { mi[r] = -1e30f; li[r] = 0.0f; }

    for (int j = 0; j <= qt; j += 2) {
        const int k0 = j * 16;               // first key of this 32-key step
        const bool have1 = (j + 1) <= qt;

        // ---- scores: S = (Q K^T) / sqrt(dk) ----
        const _Float16* Krow0 = Khead + (size_t)(k0 + ln) * DK;
        v8f s0 = {};
        s0 = wmma_f16(qa0, load_a16_f16(Krow0 + kb), s0);
        s0 = wmma_f16(qa1, load_a16_f16(Krow0 + 32 + kb), s0);
        v8f s1 = {};
        if (have1) {
            const _Float16* Krow1 = Khead + (size_t)(k0 + 16 + ln) * DK;
            s1 = wmma_f16(qa0, load_a16_f16(Krow1 + kb), s1);
            s1 = wmma_f16(qa1, load_a16_f16(Krow1 + 32 + kb), s1);
        }

        // ---- scale + causal mask + online softmax (C layout: col=ln, rows across halves)
        float scale[8];
#pragma unroll
        for (int r = 0; r < 8; ++r) {
            const int row = q0 + rbase + r;                 // absolute query index
            float v0 = s0[r] * 0.125f;                      // 1/sqrt(64)
            float v1 = s1[r] * 0.125f;
            if (k0 + ln > row)      v0 = -1e30f;
            if (k0 + 16 + ln > row) v1 = -1e30f;            // also kills absent tile1
            float vmax = fmaxf(v0, v1);
            vmax = fmaxf(vmax, __shfl_xor(vmax, 1));
            vmax = fmaxf(vmax, __shfl_xor(vmax, 2));
            vmax = fmaxf(vmax, __shfl_xor(vmax, 4));
            vmax = fmaxf(vmax, __shfl_xor(vmax, 8));
            const float mnew = fmaxf(mi[r], vmax);
            const float sc   = __expf(mi[r] - mnew);
            const float p0   = __expf(v0 - mnew);
            const float p1   = __expf(v1 - mnew);
            float rs = p0 + p1;
            rs += __shfl_xor(rs, 1);
            rs += __shfl_xor(rs, 2);
            rs += __shfl_xor(rs, 4);
            rs += __shfl_xor(rs, 8);
            li[r] = li[r] * sc + rs;
            mi[r] = mnew;
            scale[r] = sc;
            s0[r] = p0;
            s1[r] = p1;
        }
#pragma unroll
        for (int r = 0; r < 8; ++r) {
            o[0][r] *= scale[r]; o[1][r] *= scale[r];
            o[2][r] *= scale[r]; o[3][r] *= scale[r];
        }

        // ---- P: C layout -> LDS -> A layout (16x32 f16) ----
#pragma unroll
        for (int r = 0; r < 8; ++r) {
            const int row = rbase + r;
            Pl[row * 32 + ln]      = (_Float16)s0[r];
            Pl[row * 32 + 16 + ln] = (_Float16)s1[r];
        }
        __syncthreads();   // single-wave WG: lowers to LDS waits (+barrier nop)
        const v16h pa = load_a16_f16(&Pl[ln * 32 + kb]);
        __syncthreads();

        // ---- O += P * V : B-matrix from V^T (keys contiguous along K-dim) ----
#pragma unroll
        for (int t = 0; t < 4; ++t) {
            const _Float16* Vrow = Vhead + (size_t)(t * 16 + ln) * SEQ + k0;
            o[t] = wmma_f16(pa, load_a16_f16(Vrow + kb), o[t]);
        }
    }

    // ---- normalize and write AO[b, row, h*64 + d] as f16 ----
    float inv[8];
#pragma unroll
    for (int r = 0; r < 8; ++r) inv[r] = 1.0f / li[r];
#pragma unroll
    for (int t = 0; t < 4; ++t) {
#pragma unroll
        for (int r = 0; r < 8; ++r) {
            const int row = q0 + rbase + r;
            AO[((size_t)(b * SEQ + row) << 10) + h * DK + t * 16 + ln] =
                (_Float16)(o[t][r] * inv[r]);
        }
    }
}

extern "C" void kernel_launch(void* const* d_in, const int* in_sizes, int n_in,
                              void* d_out, int out_size, void* d_ws, size_t ws_size,
                              hipStream_t stream) {
    const float* x  = (const float*)d_in[0];
    const float* qw = (const float*)d_in[1];
    const float* kw = (const float*)d_in[2];
    const float* vw = (const float*)d_in[3];
    const float* ow = (const float*)d_in[4];
    float* out = (float*)d_out;

    // workspace layout (40MB):
    //  [ 0, 8MB)  xh  (f16 x)      -- reused as AO after projections complete
    //  [ 8,16MB)  qwh,kwh,vwh,owh  (f16 weights, 2MB each)
    //  [16,24MB)  Qh   [B,H,S,DK]
    //  [24,32MB)  Kh   [B,H,S,DK]
    //  [32,40MB)  VT   [B,H,DK,S]
    const size_t XSZ = (size_t)BATCH * SEQ * DMODEL * sizeof(_Float16);   // 8MB
    const size_t WSZ = (size_t)DMODEL * DMODEL * sizeof(_Float16);        // 2MB
    char* w = (char*)d_ws;
    _Float16* xh  = (_Float16*)(w);
    _Float16* AO  = (_Float16*)(w);                 // alias: x dead after projections
    _Float16* qwh = (_Float16*)(w + XSZ);
    _Float16* kwh = (_Float16*)(w + XSZ + WSZ);
    _Float16* vwh = (_Float16*)(w + XSZ + 2 * WSZ);
    _Float16* owh = (_Float16*)(w + XSZ + 3 * WSZ);
    _Float16* Qh  = (_Float16*)(w + XSZ + 4 * WSZ);
    _Float16* Kh  = (_Float16*)(w + 2 * XSZ + 4 * WSZ);
    _Float16* VT  = (_Float16*)(w + 3 * XSZ + 4 * WSZ);

    // convert inputs to f16 once (bandwidth-trivial)
    const int nx8 = (BATCH * SEQ * DMODEL) / 8;     // 524288
    const int nw8 = (DMODEL * DMODEL) / 8;          // 131072
    cvt_f32_f16<<<dim3((nx8 + 255) / 256), dim3(256), 0, stream>>>(x, xh, nx8);
    cvt_f32_f16<<<dim3((nw8 + 255) / 256), dim3(256), 0, stream>>>(qw, qwh, nw8);
    cvt_f32_f16<<<dim3((nw8 + 255) / 256), dim3(256), 0, stream>>>(kw, kwh, nw8);
    cvt_f32_f16<<<dim3((nw8 + 255) / 256), dim3(256), 0, stream>>>(vw, vwh, nw8);
    cvt_f32_f16<<<dim3((nw8 + 255) / 256), dim3(256), 0, stream>>>(ow, owh, nw8);

    // projections: one wave per 32x64 tile
    dim3 gg(DMODEL / 64, (BATCH * SEQ) / 32);       // (16, 128)
    dim3 bb(32);
    gemm16<<<gg, bb, 0, stream>>>(xh, qwh, Qh, nullptr, 0);
    gemm16<<<gg, bb, 0, stream>>>(xh, kwh, Kh, nullptr, 0);
    gemm16<<<gg, bb, 0, stream>>>(xh, vwh, VT, nullptr, 1);

    attn_kernel<<<dim3(BATCH * NHEAD * (SEQ / 16)), bb, 0, stream>>>(Qh, Kh, VT, AO);

    gemm16<<<gg, bb, 0, stream>>>(AO, owh, nullptr, out, 2);
}
